// Relu2Attention_7687991459987
// MI455X (gfx1250) — compile-verified, matrix operand order checked by reading, not run on hardware
//
#include <hip/hip_runtime.h>

typedef __attribute__((ext_vector_type(16))) __bf16 v16bf;
typedef __attribute__((ext_vector_type(8)))  float  v8f;
typedef __attribute__((ext_vector_type(4)))  unsigned int u32x4;
typedef __attribute__((ext_vector_type(8)))  int i32x8;
typedef __attribute__((ext_vector_type(4)))  int i32x4;
typedef unsigned short u16;

#if __has_builtin(__builtin_amdgcn_tensor_load_to_lds)
#define HAVE_TDM 1
#else
#define HAVE_TDM 0
#endif

union FragBF { v16bf v; u16 u[16]; uint4 q[2]; };

__device__ __forceinline__ u16 f2bf(float f) {
  __bf16 h = (__bf16)f;
  return __builtin_bit_cast(u16, h);
}

__device__ __forceinline__ v8f zero8() {
  v8f z;
#pragma unroll
  for (int i = 0; i < 8; ++i) z[i] = 0.f;
  return z;
}

// A/B fragment for v_wmma_f32_16x16x32_bf16 from an LDS tile.
// 16-bit A layout (ISA 7.12.2): lane&15 = row (A) / col (B);
// lanes<16 hold K[0..7],K[16..23]; lanes>=16 hold K[8..15],K[24..31].
__device__ __forceinline__ v16bf ld_frag(const u16* base, int row, int stride, int lane) {
  FragBF f;
  const u16* p = base + row * stride + ((lane >> 4) << 3);
  f.q[0] = *(const uint4*)(p);
  f.q[1] = *(const uint4*)(p + 16);
  return f.v;
}

__device__ __forceinline__ v8f wmma_bf16(v16bf a, v16bf b, v8f c) {
  return __builtin_amdgcn_wmma_f32_16x16x32_bf16(
      /*neg_a=*/false, a, /*neg_b=*/false, b,
      /*c_mod=*/(short)0, c, /*reuse_a=*/false, /*reuse_b=*/false);
}

#if HAVE_TDM
// TDM 2D tile load: global (strided rows of bf16) -> LDS (packed row-major).
// Descriptor per CDNA5 ISA ch.8: group0 = {count/flags, lds_addr, global_addr,
// type=2}; group1 = {data_size=2B, tensor dims (huge: no OOB clip),
// tile_dim0/1, tensor_dim0_stride}. Groups 2/3 zero (<=2D tensor).
__device__ __forceinline__ void tdm_load_2d(const void* lds_ptr, const void* gptr,
                                            unsigned int tile_w, unsigned int tile_h,
                                            unsigned int stride_elems) {
  const unsigned long long ga = (unsigned long long)(uintptr_t)gptr;
  u32x4 g0;
  g0[0] = 1u;                                            // count=1, user mode
  g0[1] = (unsigned int)(uintptr_t)lds_ptr;              // lds_addr (low 32 = LDS offset)
  g0[2] = (unsigned int)ga;                              // global_addr[31:0]
  g0[3] = (unsigned int)((ga >> 32) & 0x01FFFFFFu) | (2u << 30);  // addr[56:32], type=2
  const unsigned int td0 = 1u << 22, td1 = 1u << 22;     // generous tensor dims
  i32x8 g1;
  g1[0] = (int)(1u << 16);                               // wg_mask=0, data_size=1 (2 bytes)
  g1[1] = (int)((td0 & 0xFFFFu) << 16);                  // tensor_dim0[15:0]
  g1[2] = (int)((td0 >> 16) | ((td1 & 0xFFFFu) << 16));  // td0[31:16] | td1[15:0]
  g1[3] = (int)((td1 >> 16) | (tile_w << 16));           // td1[31:16] | tile_dim0
  g1[4] = (int)(tile_h & 0xFFFFu);                       // tile_dim1 (tile_dim2=0)
  g1[5] = (int)stride_elems;                             // tensor_dim0_stride[31:0]
  g1[6] = 0;
  g1[7] = 0;
  i32x4 z4; z4[0] = 0; z4[1] = 0; z4[2] = 0; z4[3] = 0;
#if defined(__clang_major__) && (__clang_major__ >= 23)
  i32x8 z8;
#pragma unroll
  for (int i = 0; i < 8; ++i) z8[i] = 0;
  __builtin_amdgcn_tensor_load_to_lds(g0, g1, z4, z4, z8, 0);
#else
  __builtin_amdgcn_tensor_load_to_lds(g0, g1, z4, z4, 0);
#endif
}
#endif  // HAVE_TDM

// -------------------------------------------------------------------------
// Kernel 1: qkv = x @ w_attn^T  (M=16384, N=6144, K=2048)
// Both tiles need fp32->bf16 conversion, so fills stay manual (VALU cvt).
// -------------------------------------------------------------------------
__global__ __launch_bounds__(256) void qkv_gemm_kernel(
    const float* __restrict__ x, const float* __restrict__ w_attn,
    u16* __restrict__ qb, u16* __restrict__ kb, u16* __restrict__ vb) {
  extern __shared__ char smem[];
  u16* xs  = (u16*)smem;        // [128][32] bf16
  u16* wsm = xs + 128 * 32;     // [128][32] bf16

  const int tid  = threadIdx.x;
  const int lane = tid & 31;
  const int wave = tid >> 5;
  const int wm   = wave & 3;
  const int wn   = wave >> 2;
  const int bm   = blockIdx.x * 128;
  const int bn   = blockIdx.y * 128;

  v8f acc[2][4];
#pragma unroll
  for (int i = 0; i < 2; ++i)
#pragma unroll
    for (int j = 0; j < 4; ++j) acc[i][j] = zero8();

  const int lrow = tid >> 1;
  const int lk   = (tid & 1) * 16;
  const float* srcx = x      + (size_t)(bm + lrow) * 2048 + lk;
  const float* srcw = w_attn + (size_t)(bn + lrow) * 2048 + lk;

  for (int k0 = 0; k0 < 2048; k0 += 32) {
    u16 tmp[16];
#pragma unroll
    for (int c = 0; c < 4; ++c) {
      float4 f = *(const float4*)(srcx + k0 + c * 4);
      tmp[c*4+0] = f2bf(f.x); tmp[c*4+1] = f2bf(f.y);
      tmp[c*4+2] = f2bf(f.z); tmp[c*4+3] = f2bf(f.w);
    }
    { uint4* d = (uint4*)(xs + lrow * 32 + lk); d[0] = ((uint4*)tmp)[0]; d[1] = ((uint4*)tmp)[1]; }
#pragma unroll
    for (int c = 0; c < 4; ++c) {
      float4 f = *(const float4*)(srcw + k0 + c * 4);
      tmp[c*4+0] = f2bf(f.x); tmp[c*4+1] = f2bf(f.y);
      tmp[c*4+2] = f2bf(f.z); tmp[c*4+3] = f2bf(f.w);
    }
    { uint4* d = (uint4*)(wsm + lrow * 32 + lk); d[0] = ((uint4*)tmp)[0]; d[1] = ((uint4*)tmp)[1]; }

    if (k0 + 32 < 2048) {
      __builtin_prefetch(srcx + k0 + 32, 0, 1);
      __builtin_prefetch(srcw + k0 + 32, 0, 1);
    }
    __syncthreads();

    v16bf a0 = ld_frag(xs, wm * 32 +      (lane & 15), 32, lane);
    v16bf a1 = ld_frag(xs, wm * 32 + 16 + (lane & 15), 32, lane);
#pragma unroll
    for (int cn = 0; cn < 4; ++cn) {
      v16bf bf = ld_frag(wsm, wn * 64 + cn * 16 + (lane & 15), 32, lane);
      acc[0][cn] = wmma_bf16(a0, bf, acc[0][cn]);
      acc[1][cn] = wmma_bf16(a1, bf, acc[1][cn]);
    }
    __syncthreads();
  }

  const int rowoff = (lane >> 4) << 3;
#pragma unroll
  for (int rm = 0; rm < 2; ++rm) {
#pragma unroll
    for (int cn = 0; cn < 4; ++cn) {
      const int n0  = bn + wn * 64 + cn * 16;
      const int m0  = bm + wm * 32 + rm * 16 + rowoff;
      const int col = n0 + (lane & 15);
      u16* dstbuf; int lc; float scale;
      if (n0 < 2048)      { dstbuf = qb; lc = col;        scale = 1.f / 128.f; }
      else if (n0 < 4096) { dstbuf = kb; lc = col - 2048; scale = 1.f; }
      else                { dstbuf = vb; lc = col - 4096; scale = 1.f; }
#pragma unroll
      for (int r = 0; r < 8; ++r)
        dstbuf[(size_t)(m0 + r) * 2048 + lc] = f2bf(acc[rm][cn][r] * scale);
    }
  }
}

// -------------------------------------------------------------------------
// Kernel 2: blocked local attention. One block per (b, window).
// Phase 1: S = q_blk(128xD) @ k2(256xD)^T with double-buffered TDM tile
//          loads (wave 0 drives the DMA, s_wait_tensorcnt + barrier sync),
//          mask + relu^2 -> LDS bf16.
// Phase 2: O = A(128x256) @ v2(256xD), V transposed into LDS manually.
// LDS map (bytes): [0,64K) attn_s; [64K,128K) phase1 double buffers
//                  (qs 8K + ks 16K per buffer) / phase2 vs (64K).
// -------------------------------------------------------------------------
__global__ __launch_bounds__(256) void attn_kernel(
    const u16* __restrict__ qb, const u16* __restrict__ kb,
    const u16* __restrict__ vb, u16* __restrict__ ao) {
  extern __shared__ char smem[];
  u16* attn_s = (u16*)smem;                 // [128][256] bf16
  u16* region = attn_s + 128 * 256;
  u16* vs     = region;                     // [128][256] phase 2

  const int tid  = threadIdx.x;
  const int lane = tid & 31;
  const int wave = tid >> 5;
  const int wm   = wave & 3;
  const int wn   = wave >> 2;
  const int wb   = blockIdx.x & 63;
  const int b    = blockIdx.x >> 6;
  const size_t row0 = (size_t)b * 8192 + (size_t)wb * 128;
  // 256 contiguous k/v rows starting one block before row0 (garbage when
  // wb==0; those scores are masked to zero below, inside d_ws either way).
  const long long krow0 = (long long)row0 - 128;

  // ---------------- phase 1: scores ----------------
  v8f acc[2][8];
#pragma unroll
  for (int i = 0; i < 2; ++i)
#pragma unroll
    for (int j = 0; j < 8; ++j) acc[i][j] = zero8();

#if HAVE_TDM
  u16* qsb[2] = { region,          region + 12288 };
  u16* ksb[2] = { region + 4096,   region + 12288 + 4096 };
  if (wave == 0) {
    tdm_load_2d(qsb[0], qb + row0 * 2048, 32, 128, 2048);
    tdm_load_2d(ksb[0], kb + krow0 * 2048, 32, 256, 2048);
  }
#else
  u16* qsb[1] = { region };
  u16* ksb[1] = { region + 4096 };
#endif

  for (int it = 0; it < 64; ++it) {
    const int k0 = it * 32;
#if HAVE_TDM
    const int cur = it & 1;
    if (wave == 0) __builtin_amdgcn_s_wait_tensorcnt(0);
    __syncthreads();                       // tile[cur] ready for all waves
    if ((wave == 0) && (it + 1 < 64)) {    // DMA next tile over the WMMAs
      tdm_load_2d(qsb[cur ^ 1], qb + row0 * 2048 + k0 + 32, 32, 128, 2048);
      tdm_load_2d(ksb[cur ^ 1], kb + krow0 * 2048 + k0 + 32, 32, 256, 2048);
    }
    const u16* qs = qsb[cur];
    const u16* ks = ksb[cur];
#else
    {  // q tile: 128x32, half-row per thread
      const int r_ = tid >> 1, h = (tid & 1) * 16;
      const uint4* s = (const uint4*)(qb + (row0 + r_) * 2048 + k0 + h);
      uint4* d = (uint4*)(qsb[0] + r_ * 32 + h);
      d[0] = s[0]; d[1] = s[1];
    }
    {  // k2 tile: 256x32, one row per thread
      const int j = tid;
      const bool valid = (j >= 128) || (wb > 0);
      uint4* d = (uint4*)(ksb[0] + j * 32);
      if (valid) {
        const uint4* s = (const uint4*)(kb + (krow0 + j) * 2048 + k0);
        d[0] = s[0]; d[1] = s[1]; d[2] = s[2]; d[3] = s[3];
      } else {
        uint4 z; z.x = z.y = z.z = z.w = 0u;
        d[0] = z; d[1] = z; d[2] = z; d[3] = z;
      }
    }
    __syncthreads();
    const u16* qs = qsb[0];
    const u16* ks = ksb[0];
#endif

    v16bf a0 = ld_frag(qs, wm * 32 +      (lane & 15), 32, lane);
    v16bf a1 = ld_frag(qs, wm * 32 + 16 + (lane & 15), 32, lane);
#pragma unroll
    for (int cn = 0; cn < 8; ++cn) {
      v16bf bf = ld_frag(ks, wn * 128 + cn * 16 + (lane & 15), 32, lane);
      acc[0][cn] = wmma_bf16(a0, bf, acc[0][cn]);
      acc[1][cn] = wmma_bf16(a1, bf, acc[1][cn]);
    }
    __syncthreads();                       // readers done before next DMA lands
  }

  // mask + relu^2 -> attn_s (bf16)
  const int rowoff = (lane >> 4) << 3;
#pragma unroll
  for (int rm = 0; rm < 2; ++rm) {
#pragma unroll
    for (int cn = 0; cn < 8; ++cn) {
      const int i0 = wm * 32 + rm * 16 + rowoff;
      const int j  = wn * 128 + cn * 16 + (lane & 15);
#pragma unroll
      for (int r = 0; r < 8; ++r) {
        const int i = i0 + r;
        const float s = acc[rm][cn][r];
        // prev block half: keep if i <= j (and prev exists);
        // current block half: keep if i >= j-128 (causal)
        const bool keep = (j < 128) ? ((wb > 0) && (i <= j)) : (i >= j - 128);
        const float val = (keep && s > 0.f) ? s * s : 0.f;
        attn_s[i * 256 + j] = f2bf(val);
      }
    }
  }
  __syncthreads();

  // ---------------- phase 2: out = A @ v2 ----------------
  for (int d0 = 0; d0 < 2048; d0 += 128) {
    __syncthreads();  // prior iteration's vs reads complete
    // vs[d][j] transposed fill: 128 x 256 bf16
#pragma unroll
    for (int it = 0; it < 16; ++it) {
      const int task = tid + 256 * it;
      const int j  = task & 255;
      const int dg = task >> 8;
      const bool valid = (j >= 128) || (wb > 0);
      u16 tmp[8];
      if (valid) {
        *(uint4*)tmp = *(const uint4*)(vb + (krow0 + j) * 2048 + d0 + dg * 8);
      } else {
#pragma unroll
        for (int r = 0; r < 8; ++r) tmp[r] = 0;
      }
#pragma unroll
      for (int r = 0; r < 8; ++r) vs[(dg * 8 + r) * 256 + j] = tmp[r];
    }
    __syncthreads();

    v8f oacc[2][4];
#pragma unroll
    for (int i = 0; i < 2; ++i)
#pragma unroll
      for (int j2 = 0; j2 < 4; ++j2) oacc[i][j2] = zero8();

#pragma unroll
    for (int j0 = 0; j0 < 256; j0 += 32) {
      v16bf a0 = ld_frag(attn_s + j0, wm * 32 +      (lane & 15), 256, lane);
      v16bf a1 = ld_frag(attn_s + j0, wm * 32 + 16 + (lane & 15), 256, lane);
#pragma unroll
      for (int cn = 0; cn < 4; ++cn) {
        v16bf bf = ld_frag(vs + j0, wn * 64 + cn * 16 + (lane & 15), 256, lane);
        oacc[0][cn] = wmma_bf16(a0, bf, oacc[0][cn]);
        oacc[1][cn] = wmma_bf16(a1, bf, oacc[1][cn]);
      }
    }

#pragma unroll
    for (int rm = 0; rm < 2; ++rm) {
#pragma unroll
      for (int cn = 0; cn < 4; ++cn) {
        const int m0 = wm * 32 + rm * 16 + rowoff;
        const int dc = d0 + wn * 64 + cn * 16 + (lane & 15);
#pragma unroll
        for (int r = 0; r < 8; ++r)
          ao[(row0 + m0 + r) * 2048 + dc] = f2bf(oacc[rm][cn][r]);
      }
    }
  }
}

// -------------------------------------------------------------------------
// Kernel 3: out = attn_out @ w_o^T  (M=16384, N=2048, K=2048), fp32 out.
// A tile (bf16) via TDM overlapped with the w_o fp32->bf16 converting fill.
// -------------------------------------------------------------------------
__global__ __launch_bounds__(256) void out_gemm_kernel(
    const u16* __restrict__ ao, const float* __restrict__ w_o,
    float* __restrict__ out) {
  extern __shared__ char smem[];
  u16* as_ = (u16*)smem;        // [128][32]
  u16* bs  = as_ + 128 * 32;    // [128][32]

  const int tid  = threadIdx.x;
  const int lane = tid & 31;
  const int wave = tid >> 5;
  const int wm   = wave & 3;
  const int wn   = wave >> 2;
  const int bm   = blockIdx.x * 128;
  const int bn   = blockIdx.y * 128;

  v8f acc[2][4];
#pragma unroll
  for (int i = 0; i < 2; ++i)
#pragma unroll
    for (int j = 0; j < 4; ++j) acc[i][j] = zero8();

  const int lrow = tid >> 1;
  const int lk   = (tid & 1) * 16;
  const u16*   srca = ao  + (size_t)(bm + lrow) * 2048 + lk;
  const float* srcb = w_o + (size_t)(bn + lrow) * 2048 + lk;

  for (int k0 = 0; k0 < 2048; k0 += 32) {
#if HAVE_TDM
    if (wave == 0)  // DMA the bf16 A tile while everyone converts B
      tdm_load_2d(as_, ao + (size_t)bm * 2048 + k0, 32, 128, 2048);
#else
    {
      const uint4* s = (const uint4*)(srca + k0);
      uint4* d = (uint4*)(as_ + lrow * 32 + lk);
      d[0] = s[0]; d[1] = s[1];
    }
#endif
    {  // B tile: w_o fp32 -> bf16
      u16 tmp[16];
#pragma unroll
      for (int c = 0; c < 4; ++c) {
        float4 f = *(const float4*)(srcb + k0 + c * 4);
        tmp[c*4+0] = f2bf(f.x); tmp[c*4+1] = f2bf(f.y);
        tmp[c*4+2] = f2bf(f.z); tmp[c*4+3] = f2bf(f.w);
      }
      uint4* d = (uint4*)(bs + lrow * 32 + lk);
      d[0] = ((uint4*)tmp)[0]; d[1] = ((uint4*)tmp)[1];
    }
    if (k0 + 32 < 2048) {
      __builtin_prefetch(srca + k0 + 32, 0, 1);
      __builtin_prefetch(srcb + k0 + 32, 0, 1);
    }
#if HAVE_TDM
    if (wave == 0) __builtin_amdgcn_s_wait_tensorcnt(0);
#endif
    __syncthreads();

    v16bf a0 = ld_frag(as_, wm * 32 +      (lane & 15), 32, lane);
    v16bf a1 = ld_frag(as_, wm * 32 + 16 + (lane & 15), 32, lane);
#pragma unroll
    for (int cn = 0; cn < 4; ++cn) {
      v16bf bf = ld_frag(bs, wn * 64 + cn * 16 + (lane & 15), 32, lane);
      acc[0][cn] = wmma_bf16(a0, bf, acc[0][cn]);
      acc[1][cn] = wmma_bf16(a1, bf, acc[1][cn]);
    }
    __syncthreads();
  }

  const int rowoff = (lane >> 4) << 3;
#pragma unroll
  for (int rm = 0; rm < 2; ++rm) {
#pragma unroll
    for (int cn = 0; cn < 4; ++cn) {
      const int m0  = bm + wm * 32 + rm * 16 + rowoff;
      const int col = bn + wn * 64 + cn * 16 + (lane & 15);
#pragma unroll
      for (int r = 0; r < 8; ++r)  // never re-read: bypass caches
        __builtin_nontemporal_store(acc[rm][cn][r],
                                    &out[(size_t)(m0 + r) * 2048 + col]);
    }
  }
}

// -------------------------------------------------------------------------
extern "C" void kernel_launch(void* const* d_in, const int* in_sizes, int n_in,
                              void* d_out, int out_size, void* d_ws, size_t ws_size,
                              hipStream_t stream) {
  const float* x      = (const float*)d_in[0];   // (2, 8192, 2048)
  const float* w_attn = (const float*)d_in[1];   // (6144, 2048)
  const float* w_o    = (const float*)d_in[2];   // (2048, 2048)
  float* out = (float*)d_out;                    // (2, 8192, 2048)

  const size_t MT = (size_t)16384 * 2048;        // B*T rows x D
  u16* qb = (u16*)d_ws;                          // bf16 q (pre-scaled 1/W)
  u16* kb = qb + MT;                             // bf16 k
  u16* vb = kb + MT;                             // bf16 v
  u16* ao = qb;  // attn output reuses q rows (q fully consumed per block)

  dim3 blk(256);
  // qkv GEMM: 128 x 48 blocks, 16 KB LDS
  qkv_gemm_kernel<<<dim3(128, 48), blk, 2 * 128 * 32 * sizeof(u16), stream>>>(
      x, w_attn, qb, kb, vb);
  // attention: one block per (b, window) = 128 blocks, 128 KB LDS
  attn_kernel<<<dim3(128), blk, 2 * 128 * 256 * sizeof(u16), stream>>>(
      qb, kb, vb, ao);
  // output GEMM: 128 x 16 blocks, 16 KB LDS
  out_gemm_kernel<<<dim3(128, 16), blk, 2 * 128 * 32 * sizeof(u16), stream>>>(
      ao, w_o, out);
}
